// kNNAttention_45372034515248
// MI455X (gfx1250) — matrix operand order repaired and checked by reading, not compiled, not run on hardware
//
#include <hip/hip_runtime.h>
#include <cmath>

// ---------------------------------------------------------------------------
// kNN attention for MI455X (gfx1250, wave32, WMMA).
// B=8, N=512, C=768, H=12, hd=64, TOPK=90.
// Pipeline:
//   1) qkv_gemm   : x[4096,768] @ qkv_w^T[768,2304] -> qkv bf16 (ws)
//   2) scores_gemm: per (b,h) q[512,64] @ k^T -> scaled f32 scores (d_out attn)
//   3) topk_softmax: per-row radix-select kth(90), mask, softmax in place
//   4) av_gemm    : attn[512,512] @ v[512,64] -> ctx bf16 (ws), [B,N,H,hd]
//   5) proj_gemm  : ctx[4096,768] @ proj_w^T + proj_b -> f32 out
// Matmuls: V_WMMA_F32_16X16X32_BF16 (bf16 in, f32 accumulate).
// Pure-copy LDS fills use GLOBAL_LOAD_ASYNC_TO_LDS_B128 (ASYNCcnt path).
// ---------------------------------------------------------------------------

typedef __attribute__((ext_vector_type(16))) __bf16 v16bf;
typedef __attribute__((ext_vector_type(8)))  __bf16 v8bf;
typedef __attribute__((ext_vector_type(8)))  float  v8f;

#define NUM_HEADS 12
#define TOPK 90
#define BATCH 8
#define SEQ 512
#define CH 768
#define HD 64
#define QKV_N 2304
#define MTOT 4096  // BATCH*SEQ

#ifndef __has_builtin
#define __has_builtin(x) 0
#endif

#if __has_builtin(__builtin_amdgcn_global_load_async_to_lds_b128)
#define USE_ASYNC_LDS 1
#else
#define USE_ASYNC_LDS 0
#endif

#if USE_ASYNC_LDS
// Parameter types per the toolchain diagnostic: (v4i AS1*, v4i AS3*, Ii, Ii)
typedef __attribute__((ext_vector_type(4))) int v4i;
typedef __attribute__((address_space(1))) v4i ga_v4i;
typedef __attribute__((address_space(3))) v4i lds_v4i;
#endif

// 16-byte global -> LDS copy; async (ASYNCcnt-tracked) when available.
__device__ __forceinline__ void copy16_g2lds(unsigned short* lds_dst,
                                             const unsigned short* gsrc) {
#if USE_ASYNC_LDS
    void* g = (void*)gsrc;          // strip const
    void* l = (void*)lds_dst;
    __builtin_amdgcn_global_load_async_to_lds_b128(
        (ga_v4i*)g, (lds_v4i*)l, 0, 0);
#else
    *(uint4*)lds_dst = *(const uint4*)gsrc;
#endif
}

__device__ __forceinline__ void wait_async_lds() {
#if USE_ASYNC_LDS
#if __has_builtin(__builtin_amdgcn_s_wait_asynccnt)
    __builtin_amdgcn_s_wait_asynccnt(0);
#else
    asm volatile("s_wait_asynccnt 0" ::: "memory");
#endif
#endif
}

// Hardware bf16 convert (RNE) via clang cast — avoids the integer-op sequence.
__device__ __forceinline__ unsigned short f2bf(float f) {
    union { __bf16 h; unsigned short u; } c;
    c.h = (__bf16)f;
    return c.u;
}
__device__ __forceinline__ unsigned pk2_bf16(float a, float b) {
    union { __bf16 h[2]; unsigned u; } c;
    c.h[0] = (__bf16)a; c.h[1] = (__bf16)b;
    return c.u;
}

// A fragment: 16x32 bf16 tile, row-major [16][32] in LDS (ushort units).
// Lane L holds row M=L%16; K = (L/16)*16 + {0..15} contiguous.
__device__ __forceinline__ v16bf ldsA_frag(const unsigned short* tile, int lane) {
    int row = lane & 15;
    const v8bf* p = (const v8bf*)(tile + row * 32);
    int c = lane >> 4;                 // 0 or 1
    v8bf lo = p[2 * c];
    v8bf hi = p[2 * c + 1];
    return __builtin_shufflevector(lo, hi, 0,1,2,3,4,5,6,7,8,9,10,11,12,13,14,15);
}

// B fragment: B^T tile stored row-major [16 n][32 k]. Lane L holds col N=L%16,
// K = (L/16)*16 + {0..15} contiguous.
__device__ __forceinline__ v16bf ldsB_frag(const unsigned short* bt, int lane) {
    int col = lane & 15;
    int kb  = (lane >> 4) << 4;
    return *(const v16bf*)(bt + col * 32 + kb);
}

__device__ __forceinline__ v8f wmma_bf16(v16bf a, v16bf b, v8f c) {
    return __builtin_amdgcn_wmma_f32_16x16x32_bf16(false, a, false, b, (short)0, c, false, false);
}

// ---------------------------------------------------------------------------
// Kernel 1: qkv = x @ qkv_w^T, output bf16.
// BM=128, BN=64, BK=32. 256 threads = 8 waves (4x2), 32x32 per wave.
// ---------------------------------------------------------------------------
__global__ __launch_bounds__(256) void qkv_gemm(const float* __restrict__ X,
                                                const float* __restrict__ W,
                                                unsigned short* __restrict__ out) {
    __shared__ alignas(32) unsigned short As[128 * 32];
    __shared__ alignas(32) unsigned short Bs[64 * 32];
    const int tid = threadIdx.x, lane = tid & 31, wave = tid >> 5;
    const int m0 = blockIdx.x * 128, n0 = blockIdx.y * 64;
    const int wm = (wave >> 1) * 32, wn = (wave & 1) * 32;
    v8f acc[2][2] = {};

    for (int k0 = 0; k0 < CH; k0 += 32) {
        #pragma unroll
        for (int r = 0; r < 4; ++r) {          // A: 128x32 f32 -> bf16 (packed cvt)
            int e = (tid + r * 256) * 4;
            int row = e >> 5, col = e & 31;
            float4 xv = *(const float4*)(X + (size_t)(m0 + row) * CH + k0 + col);
            unsigned* d = (unsigned*)(As + row * 32 + col);
            d[0] = pk2_bf16(xv.x, xv.y); d[1] = pk2_bf16(xv.z, xv.w);
        }
        #pragma unroll
        for (int r = 0; r < 2; ++r) {          // B^T: 64x32 (rows of W)
            int e = (tid + r * 256) * 4;
            int row = e >> 5, col = e & 31;
            float4 wv = *(const float4*)(W + (size_t)(n0 + row) * CH + k0 + col);
            unsigned* d = (unsigned*)(Bs + row * 32 + col);
            d[0] = pk2_bf16(wv.x, wv.y); d[1] = pk2_bf16(wv.z, wv.w);
        }
        if (k0 + 32 < CH) {                    // prefetch next K tile (global_prefetch_b8)
            __builtin_prefetch(X + (size_t)(m0 + (tid & 127)) * CH + k0 + 32, 0, 1);
            __builtin_prefetch(W + (size_t)(n0 + (tid & 63))  * CH + k0 + 32, 0, 1);
        }
        __syncthreads();
        v16bf a0 = ldsA_frag(As + (wm + 0)  * 32, lane);
        v16bf a1 = ldsA_frag(As + (wm + 16) * 32, lane);
        v16bf b0 = ldsB_frag(Bs + (wn + 0)  * 32, lane);
        v16bf b1 = ldsB_frag(Bs + (wn + 16) * 32, lane);
        acc[0][0] = wmma_bf16(a0, b0, acc[0][0]);
        acc[0][1] = wmma_bf16(a0, b1, acc[0][1]);
        acc[1][0] = wmma_bf16(a1, b0, acc[1][0]);
        acc[1][1] = wmma_bf16(a1, b1, acc[1][1]);
        __syncthreads();
    }
    const int crow = (lane >> 4) * 8, ccol = lane & 15;
    #pragma unroll
    for (int i = 0; i < 2; ++i)
        #pragma unroll
        for (int j = 0; j < 2; ++j)
            #pragma unroll
            for (int v = 0; v < 8; ++v) {
                int m = m0 + wm + i * 16 + crow + v;
                int n = n0 + wn + j * 16 + ccol;
                out[(size_t)m * QKV_N + n] = f2bf(acc[i][j][v]);
            }
}

// ---------------------------------------------------------------------------
// Kernel 2: per (b,h): S = scale * q @ k^T, f32 into d_out attn region.
// grid (4, 8, 96). bf16 tiles copied via async global->LDS.
// ---------------------------------------------------------------------------
__global__ __launch_bounds__(256) void scores_gemm(const unsigned short* __restrict__ qkv,
                                                   float* __restrict__ attn) {
    __shared__ alignas(32) unsigned short As[128 * 32];
    __shared__ alignas(32) unsigned short Bs[64 * 32];
    const int tid = threadIdx.x, lane = tid & 31, wave = tid >> 5;
    const int bh = blockIdx.z, b = bh / NUM_HEADS, h = bh % NUM_HEADS;
    const int m0 = blockIdx.x * 128, n0 = blockIdx.y * 64;
    const int wm = (wave >> 1) * 32, wn = (wave & 1) * 32;
    const unsigned short* Q = qkv + (size_t)(b * SEQ) * QKV_N + h * HD;
    const unsigned short* K = Q + CH;   // k block within the 3C qkv axis
    const float scale = 0.125f;         // hd^-0.5 = 1/8
    v8f acc[2][2] = {};

    for (int d0 = 0; d0 < HD; d0 += 32) {
        #pragma unroll
        for (int r = 0; r < 2; ++r) {          // A: 128x32 bf16 (512 x 16B chunks)
            int idx = tid + r * 256;
            int row = idx >> 2, c = (idx & 3) * 8;
            copy16_g2lds(As + row * 32 + c,
                         Q + (size_t)(m0 + row) * QKV_N + d0 + c);
        }
        {                                      // B^T: 64x32 bf16 (256 x 16B chunks)
            int row = tid >> 2, c = (tid & 3) * 8;
            copy16_g2lds(Bs + row * 32 + c,
                         K + (size_t)(n0 + row) * QKV_N + d0 + c);
        }
        wait_async_lds();
        __syncthreads();
        v16bf a0 = ldsA_frag(As + (wm + 0)  * 32, lane);
        v16bf a1 = ldsA_frag(As + (wm + 16) * 32, lane);
        v16bf b0 = ldsB_frag(Bs + (wn + 0)  * 32, lane);
        v16bf b1 = ldsB_frag(Bs + (wn + 16) * 32, lane);
        acc[0][0] = wmma_bf16(a0, b0, acc[0][0]);
        acc[0][1] = wmma_bf16(a0, b1, acc[0][1]);
        acc[1][0] = wmma_bf16(a1, b0, acc[1][0]);
        acc[1][1] = wmma_bf16(a1, b1, acc[1][1]);
        __syncthreads();
    }
    float* S = attn + (size_t)bh * SEQ * SEQ;
    const int crow = (lane >> 4) * 8, ccol = lane & 15;
    #pragma unroll
    for (int i = 0; i < 2; ++i)
        #pragma unroll
        for (int j = 0; j < 2; ++j)
            #pragma unroll
            for (int v = 0; v < 8; ++v) {
                int m = m0 + wm + i * 16 + crow + v;
                int n = n0 + wn + j * 16 + ccol;
                S[(size_t)m * SEQ + n] = acc[i][j][v] * scale;
            }
}

// ---------------------------------------------------------------------------
// Kernel 3: per row of 512 scores: exact kth(90)-largest via 8-pass 4-bit radix
// select on monotonic float keys, then masked softmax, in place.
// 256 threads = 8 independent waves, one row per wave, 16 values per lane.
// ---------------------------------------------------------------------------
__global__ __launch_bounds__(256) void topk_softmax(float* __restrict__ attn,
                                                    const int* __restrict__ islast) {
    const int lane = threadIdx.x & 31;
    const int row  = blockIdx.x * 8 + (threadIdx.x >> 5);
    float* p = attn + (size_t)row * SEQ;

    float vals[16];
    unsigned keys[16];
    #pragma unroll
    for (int t = 0; t < 16; ++t) {
        float v = p[t * 32 + lane];
        vals[t] = v;
        unsigned u = __float_as_uint(v);
        keys[t] = (u & 0x80000000u) ? ~u : (u | 0x80000000u);  // order-preserving
    }

    unsigned thr_key = 0u;
    if (*islast == 0) {
        unsigned prefix = 0u, msk = 0u;
        int remaining = TOPK;
        for (int shift = 28; shift >= 0; shift -= 4) {
            unsigned dmask = msk | (0xFu << shift);
            int cum = 0, chosen = 0;
            for (int bkt = 15; bkt >= 0; --bkt) {
                unsigned want = prefix | ((unsigned)bkt << shift);
                int c = 0;
                #pragma unroll
                for (int t = 0; t < 16; ++t)
                    c += ((keys[t] & dmask) == want) ? 1 : 0;
                #pragma unroll
                for (int off = 16; off > 0; off >>= 1)
                    c += __shfl_xor(c, off);
                if (cum + c >= remaining) { chosen = bkt; break; }
                cum += c;
            }
            prefix |= (unsigned)chosen << shift;
            msk = dmask;
            remaining -= cum;
        }
        thr_key = prefix;      // kth-largest key; keep key >= thr_key (ties kept)
    }

    float mx = -3.4e38f;
    #pragma unroll
    for (int t = 0; t < 16; ++t)
        if (keys[t] >= thr_key) mx = fmaxf(mx, vals[t]);
    #pragma unroll
    for (int off = 16; off > 0; off >>= 1)
        mx = fmaxf(mx, __shfl_xor(mx, off));

    float e[16];
    float s = 0.f;
    #pragma unroll
    for (int t = 0; t < 16; ++t) {
        e[t] = (keys[t] >= thr_key) ? __expf(vals[t] - mx) : 0.f;
        s += e[t];
    }
    #pragma unroll
    for (int off = 16; off > 0; off >>= 1)
        s += __shfl_xor(s, off);
    float inv = 1.f / s;
    #pragma unroll
    for (int t = 0; t < 16; ++t)
        p[t * 32 + lane] = e[t] * inv;
}

// ---------------------------------------------------------------------------
// Kernel 4: per (b,h): ctx = attn @ v -> bf16 ctx in [B,N,H,hd] layout.
// grid (4, 96): m-blocks of 128, full N=64 (=hd), K=512 in steps of 32.
// ---------------------------------------------------------------------------
__global__ __launch_bounds__(256) void av_gemm(const float* __restrict__ attn,
                                               const unsigned short* __restrict__ qkv,
                                               unsigned short* __restrict__ ctx) {
    __shared__ alignas(32) unsigned short As[128 * 32];
    __shared__ alignas(32) unsigned short Bs[64 * 32];
    const int tid = threadIdx.x, lane = tid & 31, wave = tid >> 5;
    const int bh = blockIdx.y, b = bh / NUM_HEADS, h = bh % NUM_HEADS;
    const int m0 = blockIdx.x * 128;
    const int wm = (wave >> 1) * 32, wn = (wave & 1) * 32;
    const float* A = attn + (size_t)bh * SEQ * SEQ;
    const unsigned short* V = qkv + (size_t)(b * SEQ) * QKV_N + 2 * CH + h * HD;
    v8f acc[2][2] = {};

    for (int k0 = 0; k0 < SEQ; k0 += 32) {
        #pragma unroll
        for (int r = 0; r < 4; ++r) {          // A: attn 128x32 f32 -> bf16
            int e = (tid + r * 256) * 4;
            int row = e >> 5, col = e & 31;
            float4 av = *(const float4*)(A + (size_t)(m0 + row) * SEQ + k0 + col);
            unsigned* d = (unsigned*)(As + row * 32 + col);
            d[0] = pk2_bf16(av.x, av.y); d[1] = pk2_bf16(av.z, av.w);
        }
        #pragma unroll
        for (int r = 0; r < 4; ++r) {          // B^T[d][j] from v[j][d] (LDS transpose)
            int u = tid + r * 256;             // 1024 uints: j = u>>5, ud = u&31
            int j = u >> 5, ud = u & 31;
            unsigned int w = *(const unsigned int*)(V + (size_t)(k0 + j) * QKV_N + ud * 2);
            Bs[(ud * 2 + 0) * 32 + j] = (unsigned short)(w & 0xffffu);
            Bs[(ud * 2 + 1) * 32 + j] = (unsigned short)(w >> 16);
        }
        if (k0 + 32 < SEQ) {
            __builtin_prefetch(A + (size_t)(m0 + (tid & 127)) * SEQ + k0 + 32, 0, 1);
            __builtin_prefetch(V + (size_t)(k0 + 32 + (tid & 31)) * QKV_N, 0, 1);
        }
        __syncthreads();
        v16bf a0 = ldsA_frag(As + (wm + 0)  * 32, lane);
        v16bf a1 = ldsA_frag(As + (wm + 16) * 32, lane);
        v16bf b0 = ldsB_frag(Bs + (wn + 0)  * 32, lane);
        v16bf b1 = ldsB_frag(Bs + (wn + 16) * 32, lane);
        acc[0][0] = wmma_bf16(a0, b0, acc[0][0]);
        acc[0][1] = wmma_bf16(a0, b1, acc[0][1]);
        acc[1][0] = wmma_bf16(a1, b0, acc[1][0]);
        acc[1][1] = wmma_bf16(a1, b1, acc[1][1]);
        __syncthreads();
    }
    const int crow = (lane >> 4) * 8, ccol = lane & 15;
    #pragma unroll
    for (int i = 0; i < 2; ++i)
        #pragma unroll
        for (int j = 0; j < 2; ++j)
            #pragma unroll
            for (int v = 0; v < 8; ++v) {
                int m = m0 + wm + i * 16 + crow + v;  // row within (b,h)
                int n = wn + j * 16 + ccol;           // d within head
                ctx[(size_t)(b * SEQ + m) * CH + h * HD + n] = f2bf(acc[i][j][v]);
            }
}

// ---------------------------------------------------------------------------
// Kernel 5: out = ctx @ proj_w^T + proj_b, f32 output.
// ctx tile is a pure bf16 copy -> async global->LDS path.
// ---------------------------------------------------------------------------
__global__ __launch_bounds__(256) void proj_gemm(const unsigned short* __restrict__ ctx,
                                                 const float* __restrict__ W,
                                                 const float* __restrict__ bias,
                                                 float* __restrict__ out) {
    __shared__ alignas(32) unsigned short As[128 * 32];
    __shared__ alignas(32) unsigned short Bs[64 * 32];
    const int tid = threadIdx.x, lane = tid & 31, wave = tid >> 5;
    const int m0 = blockIdx.x * 128, n0 = blockIdx.y * 64;
    const int wm = (wave >> 1) * 32, wn = (wave & 1) * 32;
    v8f acc[2][2] = {};

    for (int k0 = 0; k0 < CH; k0 += 32) {
        #pragma unroll
        for (int r = 0; r < 2; ++r) {          // A: ctx bf16 copy (async)
            int idx = tid + r * 256;
            int row = idx >> 2, c = (idx & 3) * 8;
            copy16_g2lds(As + row * 32 + c,
                         ctx + (size_t)(m0 + row) * CH + k0 + c);
        }
        #pragma unroll
        for (int r = 0; r < 2; ++r) {          // B^T: proj_w rows f32 -> bf16
            int e = (tid + r * 256) * 4;
            int row = e >> 5, col = e & 31;
            float4 wv = *(const float4*)(W + (size_t)(n0 + row) * CH + k0 + col);
            unsigned* d = (unsigned*)(Bs + row * 32 + col);
            d[0] = pk2_bf16(wv.x, wv.y); d[1] = pk2_bf16(wv.z, wv.w);
        }
        if (k0 + 32 < CH)
            __builtin_prefetch(W + (size_t)(n0 + (tid & 63)) * CH + k0 + 32, 0, 1);
        wait_async_lds();
        __syncthreads();
        v16bf a0 = ldsA_frag(As + (wm + 0)  * 32, lane);
        v16bf a1 = ldsA_frag(As + (wm + 16) * 32, lane);
        v16bf b0 = ldsB_frag(Bs + (wn + 0)  * 32, lane);
        v16bf b1 = ldsB_frag(Bs + (wn + 16) * 32, lane);
        acc[0][0] = wmma_bf16(a0, b0, acc[0][0]);
        acc[0][1] = wmma_bf16(a0, b1, acc[0][1]);
        acc[1][0] = wmma_bf16(a1, b0, acc[1][0]);
        acc[1][1] = wmma_bf16(a1, b1, acc[1][1]);
        __syncthreads();
    }
    const int crow = (lane >> 4) * 8, ccol = lane & 15;
    #pragma unroll
    for (int i = 0; i < 2; ++i)
        #pragma unroll
        for (int j = 0; j < 2; ++j) {
            int n = n0 + wn + j * 16 + ccol;
            float bv = bias[n];
            #pragma unroll
            for (int v = 0; v < 8; ++v) {
                int m = m0 + wm + i * 16 + crow + v;
                out[(size_t)m * CH + n] = acc[i][j][v] + bv;
            }
        }
}

// ---------------------------------------------------------------------------
extern "C" void kernel_launch(void* const* d_in, const int* in_sizes, int n_in,
                              void* d_out, int out_size, void* d_ws, size_t ws_size,
                              hipStream_t stream) {
    const float* x      = (const float*)d_in[0];
    const float* qkv_w  = (const float*)d_in[1];
    const float* proj_w = (const float*)d_in[2];
    const float* proj_b = (const float*)d_in[3];
    const int*   islast = (const int*)d_in[4];

    float* out  = (float*)d_out;
    float* attn = out + (size_t)BATCH * SEQ * CH;           // second tuple element

    unsigned short* qkv_bf = (unsigned short*)d_ws;         // [4096, 2304] bf16
    unsigned short* ctx_bf = qkv_bf + (size_t)MTOT * QKV_N; // [4096, 768] bf16

    qkv_gemm   <<<dim3(MTOT / 128, QKV_N / 64), 256, 0, stream>>>(x, qkv_w, qkv_bf);
    scores_gemm<<<dim3(SEQ / 128, SEQ / 64, BATCH * NUM_HEADS), 256, 0, stream>>>(qkv_bf, attn);
    topk_softmax<<<dim3(BATCH * NUM_HEADS * SEQ / 8), 256, 0, stream>>>(attn, islast);
    av_gemm    <<<dim3(SEQ / 128, BATCH * NUM_HEADS), 256, 0, stream>>>(attn, qkv_bf, ctx_bf);
    proj_gemm  <<<dim3(MTOT / 128, CH / 64), 256, 0, stream>>>(ctx_bf, proj_w, proj_b, out);
}
